// TGN_86045374808295
// MI455X (gfx1250) — matrix operand order repaired and checked by reading, not compiled, not run on hardware
//
#include <hip/hip_runtime.h>

// ---------------------------------------------------------------------------
// TGN-style GAT forward for MI455X (gfx1250, wave32, WMMA)
// ---------------------------------------------------------------------------

typedef __attribute__((ext_vector_type(16))) _Float16 v16h;
typedef __attribute__((ext_vector_type(8)))  _Float16 v8h;
typedef __attribute__((ext_vector_type(8)))  float    v8f;

#define IN_DIM   128
#define HID      64
#define OUT_DIM  16
#define NEG_SLOPE 0.2f
#define TOT_DIM  (IN_DIM + HID)   // 192
#define KPAD     8                // LDS K-stride padding (halves); keeps 16B align
#define MAX_SW   (128 * (TOT_DIM + KPAD))   // 25600 halves = 51.2 KB

// ---- monotonic float<->uint key (for atomicMax on floats) -----------------
__device__ __forceinline__ unsigned f2key(float f) {
    unsigned u = __float_as_uint(f);
    return (u & 0x80000000u) ? ~u : (u | 0x80000000u);
}
__device__ __forceinline__ float key2f(unsigned k) {
    unsigned u = (k & 0x80000000u) ? (k & 0x7fffffffu) : ~k;
    return __uint_as_float(u);
}
#define NEG_INF_KEY 0x007FFFFFu   // f2key(-inf)

// ---------------------------------------------------------------------------
// Kernel 1: time encode + concat  h0 = [x | cos(t*w+b) | sin(t*w+b)]  [N,192]
// ---------------------------------------------------------------------------
__global__ void k_time_concat(const float* __restrict__ x,
                              const float* __restrict__ ts,
                              const float* __restrict__ tw,
                              const float* __restrict__ tb,
                              float* __restrict__ h0, int N) {
    int gid = blockIdx.x * blockDim.x + threadIdx.x;
    if (gid >= N * TOT_DIM) return;
    int n = gid / TOT_DIM;
    int c = gid - n * TOT_DIM;
    float v;
    if (c < IN_DIM) {
        v = x[(size_t)n * IN_DIM + c];
    } else {
        int j = c - IN_DIM;
        float t = ts[n];
        if (j < HID / 2) v = __cosf(t * tw[j] + tb[j]);
        else { j -= HID / 2; v = __sinf(t * tw[j] + tb[j]); }
    }
    h0[gid] = v;
}

// ---------------------------------------------------------------------------
// Kernel 2: WMMA GEMM  C[M,Ncols] = A[M,K] * W[K,Ncols] (+bias)(+relu)
//   - W staged f32->f16 into LDS, TRANSPOSED with padded K-stride so a lane's
//     B fragment is two contiguous ds_load_b128 (bank-conflict-free)
//   - A fragments: four float4 global loads per k-step, cvt to f16 in regs
//   - one wave per 16-row tile, NT column tiles live in accumulators
// ---------------------------------------------------------------------------
template<int NT>
__global__ void k_gemm_wmma(const float* __restrict__ A,
                            const float* __restrict__ W,
                            const float* __restrict__ bias,
                            float* __restrict__ C,
                            int M, int K, int Ncols, int relu) {
    __shared__ _Float16 sWt[MAX_SW];       // [Ncols][K + KPAD]
    const int KP = K + KPAD;
    int tid = threadIdx.x;
    int total = K * Ncols;
    for (int i = tid; i < total; i += blockDim.x) {
        int k   = i / Ncols;               // coalesced global read
        int col = i - k * Ncols;
        sWt[col * KP + k] = (_Float16)W[i];
    }
    __syncthreads();

    int wave = tid >> 5;
    int lane = tid & 31;
    int tile_m = blockIdx.x * 8 + wave;
    int row_base = tile_m * 16;
    if (row_base >= M) return;   // no further barriers below

    int lrow  = lane & 15;       // row/col within tile
    int khalf = lane >> 4;       // which half of the wave
    int arow  = row_base + lrow; if (arow > M - 1) arow = M - 1;
    const float* Arow = A + (size_t)arow * K;
    const _Float16* Bcol = sWt + (lrow /*col in tile*/) * 0; // per-tile below

    v8f acc[NT] = {};

    for (int kb = 0; kb < K; kb += 32) {
        if (kb + 32 < K) __builtin_prefetch(Arow + kb + 32, 0, 1);
        // --- A fragment: 16x32 f16; 16 consecutive K per lane (k = 16*khalf + s)
        float4 a0 = *(const float4*)(Arow + kb + khalf * 16);
        float4 a1 = *(const float4*)(Arow + kb + khalf * 16 + 4);
        float4 a2 = *(const float4*)(Arow + kb + khalf * 16 + 8);
        float4 a3 = *(const float4*)(Arow + kb + khalf * 16 + 12);
        v16h af;
        af[0]  = (_Float16)a0.x; af[1]  = (_Float16)a0.y;
        af[2]  = (_Float16)a0.z; af[3]  = (_Float16)a0.w;
        af[4]  = (_Float16)a1.x; af[5]  = (_Float16)a1.y;
        af[6]  = (_Float16)a1.z; af[7]  = (_Float16)a1.w;
        af[8]  = (_Float16)a2.x; af[9]  = (_Float16)a2.y;
        af[10] = (_Float16)a2.z; af[11] = (_Float16)a2.w;
        af[12] = (_Float16)a3.x; af[13] = (_Float16)a3.y;
        af[14] = (_Float16)a3.z; af[15] = (_Float16)a3.w;
        // --- B fragments: 16 contiguous halves per lane, two b128 LDS loads
#pragma unroll
        for (int t = 0; t < NT; ++t) {
            int col = t * 16 + lrow;
            const v8h* bp = (const v8h*)(sWt + col * KP + kb + khalf * 16);
            v8h blo = bp[0];
            v8h bhi = bp[1];
            v16h bf = __builtin_shufflevector(blo, bhi,
                        0, 1, 2, 3, 4, 5, 6, 7, 8, 9, 10, 11, 12, 13, 14, 15);
            acc[t] = __builtin_amdgcn_wmma_f32_16x16x32_f16(
                false, af, false, bf, (short)0, acc[t], false, false);
        }
    }
    (void)Bcol;

    // --- C layout: VGPR r -> M = r + 8*khalf ; N = lane&15
#pragma unroll
    for (int t = 0; t < NT; ++t) {
#pragma unroll
        for (int r = 0; r < 8; ++r) {
            int m = row_base + r + khalf * 8;
            if (m < M) {
                int col = t * 16 + lrow;
                float v = acc[t][r];
                if (bias) v += bias[col];
                if (relu) v = v > 0.f ? v : 0.f;
                C[(size_t)m * Ncols + col] = v;
            }
        }
    }
}

// ---------------------------------------------------------------------------
// Kernel 3: per-node attention logits  as/ad[n,h] = <g[n,h,:], att[h,:]>
// ---------------------------------------------------------------------------
__global__ void k_alpha(const float* __restrict__ g,
                        const float* __restrict__ att_s,
                        const float* __restrict__ att_d,
                        float* __restrict__ as_, float* __restrict__ ad_,
                        int N, int hshift) {
    int gid = blockIdx.x * blockDim.x + threadIdx.x;
    int H = 1 << hshift;
    if (gid >= N * H) return;
    int n = gid >> hshift;
    int h = gid & (H - 1);
    const float* gp = g + ((size_t)n * H + h) * HID;
    const float* sp = att_s + h * HID;
    const float* dp = att_d + h * HID;
    float a = 0.f, b = 0.f;
#pragma unroll 8
    for (int c = 0; c < HID; ++c) { float v = gp[c]; a += v * sp[c]; b += v * dp[c]; }
    as_[gid] = a; ad_[gid] = b;
}

// ---------------------------------------------------------------------------
// Fill kernels (d_ws is poisoned 0xAA before timing; we must init ourselves)
// ---------------------------------------------------------------------------
__global__ void k_fill_f32(float* p, float v, long long n) {
    long long gid = (long long)blockIdx.x * blockDim.x + threadIdx.x;
    if (gid < n) p[gid] = v;
}
__global__ void k_fill_u32(unsigned* p, unsigned v, long long n) {
    long long gid = (long long)blockIdx.x * blockDim.x + threadIdx.x;
    if (gid < n) p[gid] = v;
}

// ---------------------------------------------------------------------------
// Edge pass A: segment max of leaky_relu(as[src]+ad[dst]) per (dst, head)
// ---------------------------------------------------------------------------
__global__ void k_edge_max(const int* __restrict__ ei,
                           const float* __restrict__ as_,
                           const float* __restrict__ ad_,
                           unsigned* __restrict__ mkey,
                           int E, int N, int hshift) {
    int H = 1 << hshift;
    long long EE = (long long)E + N;
    long long gid = (long long)blockIdx.x * blockDim.x + threadIdx.x;
    if (gid >= (EE << hshift)) return;
    int e = (int)(gid >> hshift);
    int h = (int)(gid & (H - 1));
    int s, d;
    if (e < E) { s = ei[e]; d = ei[E + e]; } else { s = d = e - E; }
    float ev = as_[s * H + h] + ad_[d * H + h];
    ev = ev > 0.f ? ev : NEG_SLOPE * ev;
    atomicMax(&mkey[d * H + h], f2key(ev));
}

// ---------------------------------------------------------------------------
// Edge pass B: ex = exp(e - max[dst]);  store ex per edge;  segment-sum denom
// ---------------------------------------------------------------------------
__global__ void k_edge_expsum(const int* __restrict__ ei,
                              const float* __restrict__ as_,
                              const float* __restrict__ ad_,
                              const unsigned* __restrict__ mkey,
                              float* __restrict__ den,
                              float* __restrict__ exb,
                              int E, int N, int hshift) {
    int H = 1 << hshift;
    long long EE = (long long)E + N;
    long long gid = (long long)blockIdx.x * blockDim.x + threadIdx.x;
    if (gid >= (EE << hshift)) return;
    int e = (int)(gid >> hshift);
    int h = (int)(gid & (H - 1));
    int s, d;
    if (e < E) { s = ei[e]; d = ei[E + e]; } else { s = d = e - E; }
    float ev = as_[s * H + h] + ad_[d * H + h];
    ev = ev > 0.f ? ev : NEG_SLOPE * ev;
    float ex = __expf(ev - key2f(mkey[d * H + h]));
    exb[gid] = ex;
    atomicAdd(&den[d * H + h], ex);
}

// ---------------------------------------------------------------------------
// Edge pass C: out[dst, c0..c0+3] += alpha(e,h) * g[src, c0..c0+3]
//   thread = 4 channels (float4 gather); wave = one edge's channel band ->
//   per-edge scalars are same-address across the wave (single L2 transaction)
// ---------------------------------------------------------------------------
__global__ void k_edge_scatter(const int* __restrict__ ei,
                               const float* __restrict__ exb,
                               const float* __restrict__ den,
                               const float* __restrict__ g,
                               float* __restrict__ out,
                               int E, int N, int hshift) {
    int chshift = hshift + 6;               // CH = H*64
    int qshift  = chshift - 2;              // CH/4 float4 chunks
    int CH = 1 << chshift;
    long long EE = (long long)E + N;
    long long gid = (long long)blockIdx.x * blockDim.x + threadIdx.x;
    if (gid >= (EE << qshift)) return;
    int e  = (int)(gid >> qshift);
    int c0 = (int)(gid & ((CH >> 2) - 1)) << 2;
    int h  = c0 >> 6;
    int H  = 1 << hshift;
    int s, d;
    if (e < E) { s = ei[e]; d = ei[E + e]; } else { s = d = e - E; }
    float alpha = exb[(long long)e * H + h] / (den[d * H + h] + 1e-16f);
    float4 gv = *(const float4*)(g + (size_t)s * CH + c0);
    float* op = out + (size_t)d * CH + c0;
    atomicAdd(op + 0, alpha * gv.x);
    atomicAdd(op + 1, alpha * gv.y);
    atomicAdd(op + 2, alpha * gv.z);
    atomicAdd(op + 3, alpha * gv.w);
}

// ---------------------------------------------------------------------------
// Elementwise bias (+relu) after aggregation
// ---------------------------------------------------------------------------
__global__ void k_bias_relu(float* __restrict__ p, const float* __restrict__ bias,
                            int N, int chshift, int relu) {
    int CH = 1 << chshift;
    long long gid = (long long)blockIdx.x * blockDim.x + threadIdx.x;
    if (gid >= ((long long)N << chshift)) return;
    int c = (int)(gid & (CH - 1));
    float v = p[gid] + bias[c];
    if (relu) v = v > 0.f ? v : 0.f;
    p[gid] = v;
}

// ---------------------------------------------------------------------------
// Host launch
// ---------------------------------------------------------------------------
static inline unsigned cdiv(long long a, long long b) { return (unsigned)((a + b - 1) / b); }

extern "C" void kernel_launch(void* const* d_in, const int* in_sizes, int n_in,
                              void* d_out, int out_size, void* d_ws, size_t ws_size,
                              hipStream_t stream) {
    const float* x    = (const float*)d_in[0];
    const float* ts   = (const float*)d_in[1];
    const int*   ei   = (const int*)d_in[2];
    const float* tw   = (const float*)d_in[3];
    const float* tb   = (const float*)d_in[4];
    const float* W1   = (const float*)d_in[5];
    const float* aS1  = (const float*)d_in[6];
    const float* aD1  = (const float*)d_in[7];
    const float* b1   = (const float*)d_in[8];
    const float* W2   = (const float*)d_in[9];
    const float* aS2  = (const float*)d_in[10];
    const float* aD2  = (const float*)d_in[11];
    const float* b2   = (const float*)d_in[12];
    const float* Wc   = (const float*)d_in[13];
    const float* bc   = (const float*)d_in[14];
    float* out        = (float*)d_out;

    const int N = in_sizes[1];          // 100000
    const int E = in_sizes[2] / 2;      // 1600000
    const long long EE = (long long)E + N;

    // ---- workspace carve-up (floats) ------------------------------------
    float*    ws   = (float*)d_ws;
    float*    h0   = ws;                               // [N,192] ; reused as out1/out2
    float*    g    = h0 + (size_t)N * TOT_DIM;         // [N,128] ; reused as g2
    float*    as_  = g + (size_t)N * 128;              // [N,2]
    float*    ad_  = as_ + (size_t)N * 2;              // [N,2]
    unsigned* mkey = (unsigned*)(ad_ + (size_t)N * 2); // [N,2]
    float*    den  = (float*)mkey + (size_t)N * 2;     // [N,2]
    float*    exb  = den + (size_t)N * 2;              // [EE,2]

    const int T = 256;
    unsigned gemm_blocks = cdiv(cdiv(N, 16), 8);

    // ================= layer 0: features =================================
    k_time_concat<<<cdiv((long long)N * TOT_DIM, T), T, 0, stream>>>(x, ts, tw, tb, h0, N);

    // ================= GAT layer 1 (H=2, concat) =========================
    k_gemm_wmma<8><<<gemm_blocks, T, 0, stream>>>(h0, W1, nullptr, g, N, TOT_DIM, 128, 0);
    k_alpha<<<cdiv((long long)N * 2, T), T, 0, stream>>>(g, aS1, aD1, as_, ad_, N, 1);
    k_fill_u32<<<cdiv((long long)N * 2, T), T, 0, stream>>>(mkey, NEG_INF_KEY, (long long)N * 2);
    k_fill_f32<<<cdiv((long long)N * 2, T), T, 0, stream>>>(den, 0.f, (long long)N * 2);
    k_fill_f32<<<cdiv((long long)N * 128, T), T, 0, stream>>>(h0, 0.f, (long long)N * 128); // out1
    k_edge_max   <<<cdiv(EE << 1, T), T, 0, stream>>>(ei, as_, ad_, mkey, E, N, 1);
    k_edge_expsum<<<cdiv(EE << 1, T), T, 0, stream>>>(ei, as_, ad_, mkey, den, exb, E, N, 1);
    k_edge_scatter<<<cdiv(EE << 5, T), T, 0, stream>>>(ei, exb, den, g, h0, E, N, 1);
    k_bias_relu<<<cdiv((long long)N * 128, T), T, 0, stream>>>(h0, b1, N, 7, 1);

    // ================= GAT layer 2 (H=1, mean==identity) =================
    k_gemm_wmma<4><<<gemm_blocks, T, 0, stream>>>(h0, W2, nullptr, g, N, 128, 64, 0);
    k_alpha<<<cdiv(N, T), T, 0, stream>>>(g, aS2, aD2, as_, ad_, N, 0);
    k_fill_u32<<<cdiv(N, T), T, 0, stream>>>(mkey, NEG_INF_KEY, N);
    k_fill_f32<<<cdiv(N, T), T, 0, stream>>>(den, 0.f, N);
    k_fill_f32<<<cdiv((long long)N * 64, T), T, 0, stream>>>(h0, 0.f, (long long)N * 64); // out2
    k_edge_max   <<<cdiv(EE, T), T, 0, stream>>>(ei, as_, ad_, mkey, E, N, 0);
    k_edge_expsum<<<cdiv(EE, T), T, 0, stream>>>(ei, as_, ad_, mkey, den, exb, E, N, 0);
    k_edge_scatter<<<cdiv(EE << 4, T), T, 0, stream>>>(ei, exb, den, g, h0, E, N, 0);
    k_bias_relu<<<cdiv((long long)N * 64, T), T, 0, stream>>>(h0, b2, N, 6, 1);

    // ================= classifier ========================================
    k_gemm_wmma<1><<<gemm_blocks, T, 0, stream>>>(h0, Wc, bc, out, N, 64, OUT_DIM, 0);

    (void)n_in; (void)out_size; (void)ws_size;
}
